// LSTM_67594195304936
// MI455X (gfx1250) — matrix operand order repaired and checked by reading, not compiled, not run on hardware
//
#include <hip/hip_runtime.h>

typedef __attribute__((ext_vector_type(16))) _Float16 v16h;
typedef __attribute__((ext_vector_type(8)))  _Float16 v8h;
typedef __attribute__((ext_vector_type(8)))  float    v8f;

#define Hh    256
#define Tt    512
#define Bb    1024
#define NCLSd 10
#define BN    32      // batch columns per workgroup
#define LDSH  272     // padded LDS row stride for h (halves): 256 + 16
#define WLP   264     // padded LDS row stride for staged weights (halves): 256 + 8
#define WGATE (256 * WLP)     // halves per staged gate matrix
#define HBUF  (BN * LDSH)     // halves per h buffer

__device__ __forceinline__ float fast_sigmoid(float v) {
    return 1.0f / (1.0f + __expf(-v));
}
__device__ __forceinline__ float fast_tanh(float v) {
    float e = __expf(-2.0f * v);
    return (1.0f - e) / (1.0f + e);
}

// ---------------------------------------------------------------------------
// Kernel 0: pack the 4 recurrent weight matrices (f32, HxH row-major) into one
// f16 buffer [gate][H][H], gate order g,i,f,o. Row-major, K contiguous.
// ---------------------------------------------------------------------------
__global__ void pack_weights_f16(const float* __restrict__ Wgh,
                                 const float* __restrict__ Wih,
                                 const float* __restrict__ Wfh,
                                 const float* __restrict__ Woh,
                                 _Float16* __restrict__ wf16) {
    int i = blockIdx.x * blockDim.x + threadIdx.x;      // 0 .. 4*H*H-1
    int g = i >> 16;                                    // H*H = 65536
    int r = i & 0xFFFF;
    const float* src = (g == 0) ? Wgh : (g == 1) ? Wih : (g == 2) ? Wfh : Woh;
    wf16[i] = (_Float16)src[r];
}

// ---------------------------------------------------------------------------
// Kernel 1: the recurrent LSTM. One workgroup = 32 batch columns, 16 waves.
// Wave w owns hidden rows [16w, 16w+16) and both 16-column batch tiles.
// Gates g,i weights staged in LDS (264-half padded rows -> conflict-free
// ds_load_b128); gates f,o streamed from L2. h double-buffered in LDS f16 so
// each timestep needs only ONE workgroup barrier.
// ---------------------------------------------------------------------------
__global__ __launch_bounds__(512) void lstm_wmma_kernel(
    const float* __restrict__ x,        // [T][B]
    const _Float16* __restrict__ wf16,  // [4][H][H]
    const float* __restrict__ Wgx, const float* __restrict__ Wix,
    const float* __restrict__ Wfx, const float* __restrict__ Wox,
    const float* __restrict__ bg,  const float* __restrict__ bi,
    const float* __restrict__ bf,  const float* __restrict__ bo,
    const float* __restrict__ Whinit,
    float* __restrict__ hT_out)         // [B][H] f32, final h transposed
{
    __shared__ _Float16 wlds[2 * WGATE];     // gates g,i : [2][256][264]  (264 KB)
    __shared__ _Float16 hbuf[2 * HBUF];      // h double buffer [2][32][272] (34 KB)

    const int tid  = threadIdx.x;
    const int w    = tid >> 5;            // wave id = hidden tile jt, 0..15
    const int lane = tid & 31;
    const int lo   = lane & 15;
    const int hi   = lane >> 4;
    const int n0   = blockIdx.x * BN;     // batch offset of this workgroup

    // ---- stage gates g,i (f16) into padded LDS: 512 threads x 1 row each ----
    {
        int gl = tid >> 8;                          // 0..1
        int j  = tid & 255;
        const _Float16* src = wf16 + ((size_t)gl * Hh + j) * Hh;
        _Float16* dst = wlds + (size_t)(gl * 256 + j) * WLP;
#pragma unroll 4
        for (int k = 0; k < Hh; k += 8)
            *(v8h*)(dst + k) = *(const v8h*)(src + k);
    }

    // Per-lane j indices for the D/C tile: VGPR e <-> row j = 16w + e + 8*hi
    const int jbase = 16 * w + 8 * hi;

    // Loop-invariant per-lane x-weights and biases (C_IN == 1 -> rank-1 term)
    float wxg[8], wxi[8], wxf[8], wxo[8];
    float vbg[8], vbi[8], vbf[8], vbo[8];
#pragma unroll
    for (int e = 0; e < 8; ++e) {
        int j = jbase + e;
        wxg[e] = Wgx[j]; wxi[e] = Wix[j]; wxf[e] = Wfx[j]; wxo[e] = Wox[j];
        vbg[e] = bg[j];  vbi[e] = bi[j];  vbf[e] = bf[j];  vbo[e] = bo[j];
    }

    // Cell state and live h (f32) in registers for both batch tiles.
    float cst[2][8];
    float hcur[2][8];
#pragma unroll
    for (int nt = 0; nt < 2; ++nt)
#pragma unroll
        for (int e = 0; e < 8; ++e) { cst[nt][e] = 0.0f; hcur[nt][e] = 0.0f; }

    // h0[j][n] = Whinit[j]  -> write f16 into h buffer 0, batch-major
#pragma unroll
    for (int nt = 0; nt < 2; ++nt) {
        alignas(16) _Float16 tmp[8];
#pragma unroll
        for (int e = 0; e < 8; ++e) tmp[e] = (_Float16)Whinit[jbase + e];
        *(v8h*)(&hbuf[(nt * 16 + lo) * LDSH + jbase]) = *(const v8h*)tmp;
    }
    __syncthreads();

    // Per-lane base pointers
    const _Float16* wl_base = wlds + (size_t)(16 * w + lo) * WLP + 8 * hi;          // staged A (g,i)
    const _Float16* wg_base = wf16 + (size_t)2 * Hh * Hh
                                   + (size_t)(16 * w + lo) * Hh + 8 * hi;           // global A (f,o)
    const int hro = lo * LDSH + 8 * hi;    // B-frag read offset within a buffer (tile 0)

    int p = 0;                             // current h buffer parity
    for (int t = 0; t < Tt; ++t) {
        float xv0 = x[(size_t)t * Bb + n0 + lo];
        float xv1 = x[(size_t)t * Bb + n0 + 16 + lo];

        const _Float16* hb0 = hbuf + p * HBUF + hro;
        const _Float16* hb1 = hb0 + 16 * LDSH;

        v8f acc[4][2];
        {
            v8f z;
#pragma unroll
            for (int q = 0; q < 8; ++q) z[q] = 0.0f;
#pragma unroll
            for (int g = 0; g < 4; ++g) { acc[g][0] = z; acc[g][1] = z; }
        }

#pragma unroll 2
        for (int k0 = 0; k0 < Hh; k0 += 32) {
            // B fragments: lane holds column (nt*16+lo), K at k0+8*hi+0..7 and k0+16+8*hi+0..7
            union { v16h v; v8h h[2]; } bf0, bf1, af;
            bf0.h[0] = *(const v8h*)(hb0 + k0);
            bf0.h[1] = *(const v8h*)(hb0 + k0 + 16);
            bf1.h[0] = *(const v8h*)(hb1 + k0);
            bf1.h[1] = *(const v8h*)(hb1 + k0 + 16);

            // gate 0 (g): A from LDS
            af.h[0] = *(const v8h*)(wl_base + k0);
            af.h[1] = *(const v8h*)(wl_base + k0 + 16);
            acc[0][0] = __builtin_amdgcn_wmma_f32_16x16x32_f16(false, af.v, false, bf0.v, (short)0, acc[0][0], false, false);
            acc[0][1] = __builtin_amdgcn_wmma_f32_16x16x32_f16(false, af.v, false, bf1.v, (short)0, acc[0][1], false, false);

            // gate 1 (i): A from LDS
            af.h[0] = *(const v8h*)(wl_base + WGATE + k0);
            af.h[1] = *(const v8h*)(wl_base + WGATE + k0 + 16);
            acc[1][0] = __builtin_amdgcn_wmma_f32_16x16x32_f16(false, af.v, false, bf0.v, (short)0, acc[1][0], false, false);
            acc[1][1] = __builtin_amdgcn_wmma_f32_16x16x32_f16(false, af.v, false, bf1.v, (short)0, acc[1][1], false, false);

            // gate 2 (f): A from global (L2-resident)
            af.h[0] = *(const v8h*)(wg_base + k0);
            af.h[1] = *(const v8h*)(wg_base + k0 + 16);
            acc[2][0] = __builtin_amdgcn_wmma_f32_16x16x32_f16(false, af.v, false, bf0.v, (short)0, acc[2][0], false, false);
            acc[2][1] = __builtin_amdgcn_wmma_f32_16x16x32_f16(false, af.v, false, bf1.v, (short)0, acc[2][1], false, false);

            // gate 3 (o): A from global (L2-resident)
            af.h[0] = *(const v8h*)(wg_base + Hh * Hh + k0);
            af.h[1] = *(const v8h*)(wg_base + Hh * Hh + k0 + 16);
            acc[3][0] = __builtin_amdgcn_wmma_f32_16x16x32_f16(false, af.v, false, bf0.v, (short)0, acc[3][0], false, false);
            acc[3][1] = __builtin_amdgcn_wmma_f32_16x16x32_f16(false, af.v, false, bf1.v, (short)0, acc[3][1], false, false);
        }

        // Elementwise gate math in registers; D layout: VGPR e -> (j=jbase+e, n=nt*16+lo)
        // Writes go to the OTHER h buffer, so no barrier is needed before them.
        _Float16* hw = hbuf + (p ^ 1) * HBUF;
#pragma unroll
        for (int nt = 0; nt < 2; ++nt) {
            float xv = nt ? xv1 : xv0;
            alignas(16) _Float16 hn[8];
#pragma unroll
            for (int e = 0; e < 8; ++e) {
                float pg = acc[0][nt][e] + wxg[e] * xv + vbg[e];
                float pi = acc[1][nt][e] + wxi[e] * xv + vbi[e];
                float pf = acc[2][nt][e] + wxf[e] * xv + vbf[e];
                float po = acc[3][nt][e] + wxo[e] * xv + vbo[e];
                float gg = fast_tanh(pg);
                float ii = fast_sigmoid(pi);
                float ff = fast_sigmoid(pf);
                float oo = fast_sigmoid(po);
                float c  = gg * ii + cst[nt][e] * ff;
                cst[nt][e] = c;
                float hv = fast_tanh(c) * oo;
                hcur[nt][e] = hv;
                hn[e] = (_Float16)hv;
            }
            *(v8h*)(&hw[(nt * 16 + lo) * LDSH + jbase]) = *(const v8h*)hn;
        }
        p ^= 1;
        __syncthreads();   // single barrier per step: new h visible, old reads done
    }

    // Final h -> global, transposed [n][j] f32 for the projection kernel
#pragma unroll
    for (int nt = 0; nt < 2; ++nt) {
        int n = n0 + nt * 16 + lo;
        float* dst = hT_out + (size_t)n * Hh + jbase;
        float4 a = make_float4(hcur[nt][0], hcur[nt][1], hcur[nt][2], hcur[nt][3]);
        float4 b = make_float4(hcur[nt][4], hcur[nt][5], hcur[nt][6], hcur[nt][7]);
        *(float4*)(dst)     = a;
        *(float4*)(dst + 4) = b;
    }
}

// ---------------------------------------------------------------------------
// Kernel 2: p = Wph @ h + bp, softmax over the BATCH axis (reference axis=1),
// output y.T as [B][NCLS]. One block per class, 1024 threads = batch columns.
// ---------------------------------------------------------------------------
__global__ __launch_bounds__(1024) void proj_softmax_kernel(
    const float* __restrict__ hT,   // [B][H]
    const float* __restrict__ Wph,  // [NCLS][H]
    const float* __restrict__ bp,   // [NCLS]
    float* __restrict__ out)        // [B][NCLS]
{
    __shared__ float red[1024];
    const int c = blockIdx.x;
    const int n = threadIdx.x;

    const float* hr = hT + (size_t)n * Hh;
    const float* wr = Wph + (size_t)c * Hh;
    float acc = bp[c];
#pragma unroll 8
    for (int k = 0; k < Hh; ++k) acc += wr[k] * hr[k];

    // max over batch
    red[n] = acc;
    __syncthreads();
    for (int s = 512; s > 0; s >>= 1) {
        if (n < s) red[n] = fmaxf(red[n], red[n + s]);
        __syncthreads();
    }
    float m = red[0];
    __syncthreads();

    float e = __expf(acc - m);
    red[n] = e;
    __syncthreads();
    for (int s = 512; s > 0; s >>= 1) {
        if (n < s) red[n] += red[n + s];
        __syncthreads();
    }
    float sum = red[0];

    out[(size_t)n * NCLSd + c] = e / sum;
}

// ---------------------------------------------------------------------------
extern "C" void kernel_launch(void* const* d_in, const int* in_sizes, int n_in,
                              void* d_out, int out_size, void* d_ws, size_t ws_size,
                              hipStream_t stream) {
    const float* x      = (const float*)d_in[0];
    const float* Wgx    = (const float*)d_in[1];
    const float* Wix    = (const float*)d_in[2];
    const float* Wfx    = (const float*)d_in[3];
    const float* Wox    = (const float*)d_in[4];
    const float* Wgh    = (const float*)d_in[5];
    const float* Wih    = (const float*)d_in[6];
    const float* Wfh    = (const float*)d_in[7];
    const float* Woh    = (const float*)d_in[8];
    const float* bg     = (const float*)d_in[9];
    const float* bi     = (const float*)d_in[10];
    const float* bf     = (const float*)d_in[11];
    const float* bo     = (const float*)d_in[12];
    const float* Whinit = (const float*)d_in[13];
    const float* Wph    = (const float*)d_in[14];
    const float* bp     = (const float*)d_in[15];

    // Workspace layout: [0, 512KB) f16 packed weights; then [B][H] f32 final h.
    _Float16* wf16 = (_Float16*)d_ws;
    float*    hT   = (float*)((char*)d_ws + (size_t)4 * Hh * Hh * sizeof(_Float16));

    pack_weights_f16<<<(4 * Hh * Hh) / 256, 256, 0, stream>>>(Wgh, Wih, Wfh, Woh, wf16);

    lstm_wmma_kernel<<<Bb / BN, 512, 0, stream>>>(
        x, wf16, Wgx, Wix, Wfx, Wox, bg, bi, bf, bo, Whinit, hT);

    proj_softmax_kernel<<<NCLSd, 1024, 0, stream>>>(hT, Wph, bp, (float*)d_out);
}
